// JointEncoder_71811853189763
// MI455X (gfx1250) — matrix operand order repaired and checked by reading, not compiled, not run on hardware
//
#include <hip/hip_runtime.h>

// ---------------------------------------------------------------------------
// MI455X (gfx1250) implementation of JointEncoder reference.
// All dense GEMMs use V_WMMA_F32_16X16X32_F16 (f16 inputs, f32 accumulate).
// wave32 everywhere; LDS-staged fragments use contiguous k order (same in A and B):
//   A  (16x32 f16): lane m=lane&15, half=lane>>4, elem e: k = 2v+p+16*half
//   B  (32x16 f16): lane n=lane&15, half=lane>>4, elem e: k = 2v+p+16*half
//   C/D(16x16 f32): elem i -> row m = i + 8*half, col n = lane&15
// Weights are converted f32->f16 AND transposed to [N][K] so both GEMM panels
// load as contiguous 16-byte chunks. The panel copy uses
// GLOBAL_LOAD_ASYNC_TO_LDS_B128 (ASYNCcnt-tracked, bypasses VGPRs).
// ---------------------------------------------------------------------------

typedef __attribute__((ext_vector_type(16))) _Float16 v16h;
typedef __attribute__((ext_vector_type(8)))  float    v8f;
typedef __attribute__((ext_vector_type(8)))  _Float16 h16x8;
typedef __attribute__((ext_vector_type(4)))  _Float16 h16x4;
typedef __attribute__((ext_vector_type(2)))  _Float16 h16x2;
typedef __attribute__((ext_vector_type(4)))  float    f32x4;
typedef __attribute__((ext_vector_type(2)))  float    f32x2;

#define BSZ     32
#define LSEQ    256
#define KC      32
#define HID     256
#define HEADS   8
#define DH      32
#define NLAYERS 2
#define PROOUT  64
#define NEGBIG  (-1e9f)

// ---- gfx1250 async global->LDS path (compile-safe feature probe) ----------
#if defined(__has_builtin)
#if __has_builtin(__builtin_amdgcn_global_load_async_to_lds_b128)
#define USE_ASYNC_LDS 1
#endif
#endif

#ifdef USE_ASYNC_LDS
typedef int v4i __attribute__((vector_size(16)));
typedef __attribute__((address_space(1))) v4i* g4_ptr;  // global int4*
typedef __attribute__((address_space(3))) v4i* l4_ptr;  // LDS int4*
// integer round-trips: generic->u64->AS(1); generic LDS addr low 32 bits are
// the DS offset (aperture rule), so generic->u32->AS(3) is the DS address.
__device__ __forceinline__ void async_copy16(const _Float16* g, _Float16* l) {
    __builtin_amdgcn_global_load_async_to_lds_b128(
        (g4_ptr)(unsigned long long)g,
        (l4_ptr)(unsigned int)(unsigned long long)l,
        0, 0);
}
#endif

// ---------------------------------------------------------------------------
// helpers
// ---------------------------------------------------------------------------
__device__ __forceinline__ float block_sum_256(float v, float* red) {
    int t = threadIdx.x;
    red[t] = v;
    __syncthreads();
    for (int s = 128; s > 0; s >>= 1) {
        if (t < s) red[t] += red[t + s];
        __syncthreads();
    }
    float r = red[0];
    __syncthreads();
    return r;
}

// ---------------------------------------------------------------------------
// input transpose: src (L,BS,HID) -> x (BS,L,HID) f32 + f16 copy (4-wide)
// ---------------------------------------------------------------------------
__global__ __launch_bounds__(256) void transpose_in_kernel(
    const float* __restrict__ src, float* __restrict__ x, _Float16* __restrict__ xh) {
    int e  = (blockIdx.x * 256 + threadIdx.x) * 4;  // [0, L*BS*HID)
    int l  = e >> 13;                               // / (BS*HID)
    int r  = e & 8191;
    int b  = r >> 8;
    int hh = r & 255;
    f32x4 v = *(const f32x4*)&src[e];
    int o = (b * LSEQ + l) * HID + hh;
    *(f32x4*)&x[o] = v;
    h16x4 hv = { (_Float16)v.x, (_Float16)v.y, (_Float16)v.z, (_Float16)v.w };
    *(h16x4*)&xh[o] = hv;
}

// f32 -> f16, 4 elements per thread (n must be a multiple of 4)
__global__ __launch_bounds__(256) void f2h4_kernel(
    const float* __restrict__ in, _Float16* __restrict__ out, int n4) {
    int i = blockIdx.x * 256 + threadIdx.x;
    if (i < n4) {
        f32x4 v = ((const f32x4*)in)[i];
        h16x4 o = { (_Float16)v.x, (_Float16)v.y, (_Float16)v.z, (_Float16)v.w };
        ((h16x4*)out)[i] = o;
    }
}

// weight convert + transpose: in f32 [K][N] -> out f16 [N][K] (write-coalesced)
__global__ __launch_bounds__(256) void f2h_transpose_kernel(
    const float* __restrict__ in, _Float16* __restrict__ out, int K, int N) {
    int i = blockIdx.x * 256 + threadIdx.x;  // over N*K
    if (i < N * K) {
        int n = i / K, k = i - n * K;
        out[i] = (_Float16)in[k * N + n];
    }
}

__global__ __launch_bounds__(256) void zero_kernel(float* p, int n) {
    int i = blockIdx.x * 256 + threadIdx.x;
    if (i < n) p[i] = 0.f;
}

// ---------------------------------------------------------------------------
// WMMA GEMM: C[M,N] = A[M,K] @ Bt^T + bias (Bt is f16 [N][K]); optional relu.
// block = 256 threads (8 waves). block tile 64x64, wave tile 16x32.
// Panels: one 16-byte chunk per thread per panel, either async-to-LDS or
// global_load_b128 + ds_store_b128.
// ---------------------------------------------------------------------------
__global__ __launch_bounds__(256) void wmma_gemm_kernel(
    const _Float16* __restrict__ A, const _Float16* __restrict__ Bt,
    const float* __restrict__ bias, float* __restrict__ C,
    int M, int N, int K, int relu) {
    __shared__ __align__(16) _Float16 As[64 * 32];  // [m][k]
    __shared__ __align__(16) _Float16 Bs[64 * 32];  // [n][k]

    int t    = threadIdx.x;
    int wave = t >> 5, lane = t & 31;
    int wr   = wave >> 1, wc = wave & 1;            // 4x2 wave grid
    int nlo  = lane & 15, hf = lane >> 4;
    int m0   = blockIdx.y * 64, n0 = blockIdx.x * 64;

    int pr = t >> 2;            // panel row 0..63
    int pc = (t & 3) * 8;       // panel col (k) 0..24, 8 contiguous f16
    const int aoff = (m0 + pr) * K + pc;
    const int boff = (n0 + pr) * K + pc;

    v8f acc0 = {};
    v8f acc1 = {};

    for (int kb = 0; kb < K; kb += 32) {
#ifdef USE_ASYNC_LDS
        async_copy16(&A[aoff + kb], &As[pr * 32 + pc]);
        async_copy16(&Bt[boff + kb], &Bs[pr * 32 + pc]);
        asm volatile("s_wait_asynccnt 0x0" ::: "memory");
#else
        *(h16x8*)&As[pr * 32 + pc] = *(const h16x8*)&A[aoff + kb];
        *(h16x8*)&Bs[pr * 32 + pc] = *(const h16x8*)&Bt[boff + kb];
#endif
        if (kb + 32 < K) {
            __builtin_prefetch(&A[aoff + kb + 32], 0, 1);
            __builtin_prefetch(&Bt[boff + kb + 32], 0, 1);
        }
        __syncthreads();

        v16h a, b0, b1;
#pragma unroll
        for (int v = 0; v < 8; ++v) {
            int ak = 2 * v + 16 * hf;
            h16x2 pa = *(const h16x2*)&As[(wr * 16 + nlo) * 32 + ak];
            a[2 * v] = pa.x; a[2 * v + 1] = pa.y;
            int bk = 2 * v + 16 * hf;
            h16x2 p0 = *(const h16x2*)&Bs[(wc * 32 + nlo) * 32 + bk];
            b0[2 * v] = p0.x; b0[2 * v + 1] = p0.y;
            h16x2 p1 = *(const h16x2*)&Bs[(wc * 32 + 16 + nlo) * 32 + bk];
            b1[2 * v] = p1.x; b1[2 * v + 1] = p1.y;
        }
        acc0 = __builtin_amdgcn_wmma_f32_16x16x32_f16(false, a, false, b0, (short)0, acc0, false, false);
        acc1 = __builtin_amdgcn_wmma_f32_16x16x32_f16(false, a, false, b1, (short)0, acc1, false, false);
        __syncthreads();
    }

#pragma unroll
    for (int i = 0; i < 8; ++i) {
        int m = m0 + wr * 16 + i + 8 * hf;
        int n = n0 + wc * 32 + nlo;
        float v0 = acc0[i] + bias[n];
        float v1 = acc1[i] + bias[n + 16];
        if (relu) { v0 = fmaxf(v0, 0.f); v1 = fmaxf(v1, 0.f); }
        C[m * N + n]      = v0;
        C[m * N + n + 16] = v1;
    }
}

// ---------------------------------------------------------------------------
// fused attention per (b, h): S = Q K^T * 1/sqrt(DH), mask, softmax, ctx = P V
// block = 64 threads (2 waves); each wave owns 8 q-tiles of 16 rows.
// K, V staged in LDS as f16; softmax register-resident via shfl_xor.
// ---------------------------------------------------------------------------
__global__ __launch_bounds__(64) void attention_kernel(
    const float* __restrict__ qkv, const int* __restrict__ src_len,
    float* __restrict__ ctx) {
    __shared__ __align__(16) _Float16 Ksh[LSEQ * DH];    // [key][d]
    __shared__ __align__(16) _Float16 VshT[DH * LSEQ];   // [d][key]
    __shared__ __align__(16) _Float16 Pw[2][16 * LSEQ];  // per-wave prob rows

    int bh = blockIdx.x;
    int b  = bh >> 3, h = bh & 7;
    int t  = threadIdx.x, wave = t >> 5, lane = t & 31;
    int nlo = lane & 15, hf = lane >> 4;
    int len = src_len[b];
    const int rowbase = b * LSEQ;
    const float SC = 0.17677669529663687f;               // 1/sqrt(32)

    // stage K (row-major) and V (transposed) as f16; float2 global loads
    for (int idx = t; idx < LSEQ * DH / 2; idx += 64) {
        int key = idx >> 4, dp = (idx & 15) * 2;
        int q = (rowbase + key) * (3 * HID) + h * DH;
        f32x2 kv = *(const f32x2*)&qkv[q + HID + dp];
        h16x2 kh = { (_Float16)kv.x, (_Float16)kv.y };
        *(h16x2*)&Ksh[key * 32 + dp] = kh;
        f32x2 vv = *(const f32x2*)&qkv[q + 2 * HID + dp];
        VshT[dp * LSEQ + key]       = (_Float16)vv.x;
        VshT[(dp + 1) * LSEQ + key] = (_Float16)vv.y;
    }
    __syncthreads();

    for (int qt = wave; qt < 16; qt += 2) {
        int q0 = qt * 16;
        // Q fragment (f32 global -> f16), pair loads
        v16h qa;
        {
            const float* qp = qkv + (rowbase + q0 + nlo) * (3 * HID) + h * DH;
#pragma unroll
            for (int v = 0; v < 8; ++v) {
                int ak = 2 * v + 16 * hf;
                f32x2 pv = *(const f32x2*)&qp[ak];
                qa[2 * v]     = (_Float16)pv.x;
                qa[2 * v + 1] = (_Float16)pv.y;
            }
        }
        // S = Q K^T  (16 column tiles)
        v8f sreg[16];
#pragma unroll
        for (int j = 0; j < 16; ++j) {
            v16h bk;
#pragma unroll
            for (int v = 0; v < 8; ++v) {
                h16x2 pr = *(const h16x2*)&Ksh[(j * 16 + nlo) * 32 + 2 * v + 16 * hf];
                bk[2 * v] = pr.x; bk[2 * v + 1] = pr.y;
            }
            v8f z = {};
            sreg[j] = __builtin_amdgcn_wmma_f32_16x16x32_f16(false, qa, false, bk, (short)0, z, false, false);
        }
        // scale + mask
#pragma unroll
        for (int j = 0; j < 16; ++j)
#pragma unroll
            for (int i = 0; i < 8; ++i) {
                int qr = q0 + i + 8 * hf, kc = j * 16 + nlo;
                sreg[j][i] = (qr < len && kc < len) ? sreg[j][i] * SC : NEGBIG;
            }
        // softmax over 256 keys: 16-lane half-group reductions
        float mx[8], sm[8];
#pragma unroll
        for (int i = 0; i < 8; ++i) {
            mx[i] = sreg[0][i];
#pragma unroll
            for (int j = 1; j < 16; ++j) mx[i] = fmaxf(mx[i], sreg[j][i]);
        }
#pragma unroll
        for (int i = 0; i < 8; ++i)
            for (int d = 1; d < 16; d <<= 1) mx[i] = fmaxf(mx[i], __shfl_xor(mx[i], d, 32));
#pragma unroll
        for (int i = 0; i < 8; ++i) sm[i] = 0.f;
#pragma unroll
        for (int j = 0; j < 16; ++j)
#pragma unroll
            for (int i = 0; i < 8; ++i) {
                float e = __expf(sreg[j][i] - mx[i]);
                sreg[j][i] = e;
                sm[i] += e;
            }
#pragma unroll
        for (int i = 0; i < 8; ++i)
            for (int d = 1; d < 16; d <<= 1) sm[i] += __shfl_xor(sm[i], d, 32);
        // P rows -> LDS (re-layout D-frag -> A-frag source)
#pragma unroll
        for (int j = 0; j < 16; ++j)
#pragma unroll
            for (int i = 0; i < 8; ++i)
                Pw[wave][(i + 8 * hf) * LSEQ + j * 16 + nlo] = (_Float16)(sreg[j][i] / sm[i]);
        // ctx = P V  (two 16-wide d-subtiles, K-loop of 8x32)
#pragma unroll
        for (int nh = 0; nh < 2; ++nh) {
            v8f acc = {};
#pragma unroll
            for (int tt = 0; tt < 8; ++tt) {
                v16h pa, bv;
#pragma unroll
                for (int v = 0; v < 8; ++v) {
                    int ak = 2 * v + 16 * hf;
                    h16x2 pp = *(const h16x2*)&Pw[wave][nlo * LSEQ + tt * 32 + ak];
                    pa[2 * v] = pp.x; pa[2 * v + 1] = pp.y;
                    h16x2 pv = *(const h16x2*)&VshT[(nh * 16 + nlo) * LSEQ + tt * 32 + 2 * v + 16 * hf];
                    bv[2 * v] = pv.x; bv[2 * v + 1] = pv.y;
                }
                acc = __builtin_amdgcn_wmma_f32_16x16x32_f16(false, pa, false, bv, (short)0, acc, false, false);
            }
#pragma unroll
            for (int i = 0; i < 8; ++i)
                ctx[(rowbase + q0 + i + 8 * hf) * HID + h * DH + nh * 16 + nlo] = acc[i];
        }
    }
}

// ---------------------------------------------------------------------------
// y = LN(a + r) -> f32 + f16; one row (256) per block
// ---------------------------------------------------------------------------
__global__ __launch_bounds__(256) void add_ln_kernel(
    const float* __restrict__ a, const float* __restrict__ r,
    const float* __restrict__ g, const float* __restrict__ be,
    float* __restrict__ outf, _Float16* __restrict__ outh) {
    __shared__ float red[256];
    int row = blockIdx.x, t = threadIdx.x;
    int o = row * HID + t;
    float v = a[o] + r[o];
    float mean = block_sum_256(v, red) * (1.f / 256.f);
    float d = v - mean;
    float var = block_sum_256(d * d, red) * (1.f / 256.f);
    float y = d * rsqrtf(var + 1e-5f) * g[t] + be[t];
    outf[o] = y;
    outh[o] = (_Float16)y;
}

// ---------------------------------------------------------------------------
// fused candidate head per (b,l): comb tile in LDS, masked softmax, context,
// fusion LN, pooled atomics, sigmoid scores.
// ---------------------------------------------------------------------------
__global__ __launch_bounds__(256) void candi_fused_kernel(
    const float* __restrict__ x, const int* __restrict__ src_len,
    const int* __restrict__ candi_ids, const float* __restrict__ candi_feats,
    const int* __restrict__ candi_masks, const float* __restrict__ emb,
    const float* __restrict__ fcw, const float* __restrict__ fcb,
    const float* __restrict__ fg, const float* __restrict__ fbias,
    float* __restrict__ pool, float* __restrict__ out_fusion,
    float* __restrict__ out_mm) {
    __shared__ float comb[KC * HID];
    __shared__ float g_lds[HID], f_lds[HID], red[HID];
    __shared__ float sc[KC], aw[KC];

    int bl = blockIdx.x, b = bl >> 8, l = bl & 255;
    int t  = threadIdx.x;
    int len = src_len[b];
    const int base_blk = (b * LSEQ + l) * KC;

    for (int k = 0; k < KC; ++k) {
        int id = candi_ids[base_blk + k];
        float s = emb[id * HID + t] + fcb[t];
        const float* ft = candi_feats + (base_blk + k) * 9;
#pragma unroll
        for (int j = 0; j < 9; ++j) s += ft[j] * fcw[j * HID + t];
        comb[k * HID + t] = s;
    }
    float g = x[(b * LSEQ + l) * HID + t] * ((l < len) ? 1.f : 0.f);
    g_lds[t] = g;
    __syncthreads();

    int wave = t >> 5, lane = t & 31;
    // scores[k] = dot(gps, comb[k]) / sqrt(HID)
#pragma unroll
    for (int kk = 0; kk < 4; ++kk) {
        int k = wave * 4 + kk;
        float p = 0.f;
#pragma unroll
        for (int i = 0; i < 8; ++i) p += g_lds[lane + 32 * i] * comb[k * HID + lane + 32 * i];
        for (int d = 1; d < 32; d <<= 1) p += __shfl_xor(p, d, 32);
        if (lane == 0) sc[k] = p * 0.0625f;
    }
    __syncthreads();
    if (t < 32) {  // wave 0: masked softmax over K
        float s = (candi_masks[base_blk + t] == 0) ? NEGBIG : sc[t];
        float mx = s;
        for (int d = 1; d < 32; d <<= 1) mx = fmaxf(mx, __shfl_xor(mx, d, 32));
        float e = __expf(s - mx);
        float sum = e;
        for (int d = 1; d < 32; d <<= 1) sum += __shfl_xor(sum, d, 32);
        aw[t] = e / sum;
    }
    __syncthreads();
    float c = 0.f;
    for (int k = 0; k < KC; ++k) c += aw[k] * comb[k * HID + t];
    float f = g + c;
    float mean = block_sum_256(f, red) * (1.f / 256.f);
    float dv = f - mean;
    float var = block_sum_256(dv * dv, red) * (1.f / 256.f);
    float fn = dv * rsqrtf(var + 1e-5f) * fg[t] + fbias[t];
    f_lds[t] = fn;
    out_fusion[(l * BSZ + b) * HID + t] = fn;  // (L,BS,HID)
    if (l < len) atomicAdd(&pool[b * HID + t], fn);
    __syncthreads();
    // mm[k] = dot(fusion, comb[k]); prob = mask ? sigmoid : 0
#pragma unroll
    for (int kk = 0; kk < 4; ++kk) {
        int k = wave * 4 + kk;
        float p = 0.f;
#pragma unroll
        for (int i = 0; i < 8; ++i) p += f_lds[lane + 32 * i] * comb[k * HID + lane + 32 * i];
        for (int d = 1; d < 32; d <<= 1) p += __shfl_xor(p, d, 32);
        if (lane == 0) {
            int msk = candi_masks[base_blk + k];
            out_mm[base_blk + k] = (msk == 0) ? 0.f : 1.f / (1.f + __expf(-p));
        }
    }
}

// ---------------------------------------------------------------------------
// hidden = tanh(concat(extra, pooled/len) @ fc_hid_w + b); one batch per block
// ---------------------------------------------------------------------------
__global__ __launch_bounds__(256) void pool_gate_kernel(
    const float* __restrict__ pool, const int* __restrict__ src_len,
    const int* __restrict__ pro, const float* __restrict__ tw,
    const float* __restrict__ fw, const float* __restrict__ fb,
    float* __restrict__ out_hidden) {
    __shared__ float cat[HID + PROOUT];
    int b = blockIdx.x, t = threadIdx.x;
    if (t < PROOUT) cat[t] = tw[pro[b] * PROOUT + t];
    cat[PROOUT + t] = pool[b * HID + t] / (float)src_len[b];
    __syncthreads();
    float o = fb[t];
    for (int j = 0; j < HID + PROOUT; ++j) o += cat[j] * fw[j * HID + t];
    out_hidden[b * HID + t] = tanhf(o);
}

// ---------------------------------------------------------------------------
// host launch
// ---------------------------------------------------------------------------
extern "C" void kernel_launch(void* const* d_in, const int* in_sizes, int n_in,
                              void* d_out, int out_size, void* d_ws, size_t ws_size,
                              hipStream_t stream) {
    (void)in_sizes; (void)n_in; (void)out_size; (void)ws_size;

    const float* src         = (const float*)d_in[0];
    const int*   src_len     = (const int*)d_in[1];
    /* d_in[2] src_seg_ids: unused by reference */
    const int*   candi_ids   = (const int*)d_in[3];
    const float* candi_feats = (const float*)d_in[4];
    const int*   candi_masks = (const int*)d_in[5];
    const int*   pro         = (const int*)d_in[6];
    const float* emb         = (const float*)d_in[7];
    const float* fcw         = (const float*)d_in[8];
    const float* fcb         = (const float*)d_in[9];
    const float* fusion_g    = (const float*)d_in[10];
    const float* fusion_b    = (const float*)d_in[11];
    const float* temporal_w  = (const float*)d_in[12];
    const float* fc_hid_w    = (const float*)d_in[13];
    const float* fc_hid_b    = (const float*)d_in[14];
    const float* attn_w      = (const float*)d_in[15];
    const float* attn_b      = (const float*)d_in[16];
    const float* out_w       = (const float*)d_in[17];
    const float* out_b       = (const float*)d_in[18];
    const float* ln1_g       = (const float*)d_in[19];
    const float* ln1_b       = (const float*)d_in[20];
    const float* ffn_w1      = (const float*)d_in[21];
    const float* ffn_b1      = (const float*)d_in[22];
    const float* ffn_w2      = (const float*)d_in[23];
    const float* ffn_b2      = (const float*)d_in[24];
    const float* ln2_g       = (const float*)d_in[25];
    const float* ln2_b       = (const float*)d_in[26];

    char* ws = (char*)d_ws;
    float*    xbuf = (float*)(ws + 0);               // 8 MB   (8192x256 f32)
    _Float16* xh   = (_Float16*)(ws + 8388608);      // 4 MB
    float*    qkv  = (float*)(ws + 12582912);        // 24 MB  (8192x768 f32)
    float*    ctx  = (float*)(ws + 37748736);        // 8 MB
    _Float16* ctxh = (_Float16*)(ws + 46137344);     // 4 MB
    float*    tmp  = (float*)(ws + 50331648);        // 8 MB
    float*    h1   = (float*)(ws + 58720256);        // 32 MB  (8192x1024 f32)
    _Float16* h1h  = (_Float16*)(ws + 92274688);     // 16 MB
    _Float16* wh   = (_Float16*)(ws + 109051904);    // 512 KB weight staging (transposed)
    float*    pool = (float*)(ws + 109576192);       // 32 KB

    const int M = BSZ * LSEQ;  // 8192
    float* outF   = (float*)d_out;
    float* outHid = outF + LSEQ * BSZ * HID;
    float* outMM  = outHid + BSZ * HID;

    transpose_in_kernel<<<(LSEQ * BSZ * HID / 4) / 256, 256, 0, stream>>>(src, xbuf, xh);
    zero_kernel<<<(BSZ * HID + 255) / 256, 256, 0, stream>>>(pool, BSZ * HID);

    for (int l = 0; l < NLAYERS; ++l) {
        // QKV projection
        f2h_transpose_kernel<<<(HID * 3 * HID + 255) / 256, 256, 0, stream>>>(
            attn_w + l * HID * 3 * HID, wh, HID, 3 * HID);
        wmma_gemm_kernel<<<dim3(3 * HID / 64, M / 64), 256, 0, stream>>>(
            xh, wh, attn_b + l * 3 * HID, qkv, M, 3 * HID, HID, 0);
        // attention
        attention_kernel<<<BSZ * HEADS, 64, 0, stream>>>(qkv, src_len, ctx);
        // out projection + LN1
        f2h4_kernel<<<(M * HID / 4 + 255) / 256, 256, 0, stream>>>(ctx, ctxh, M * HID / 4);
        f2h_transpose_kernel<<<(HID * HID + 255) / 256, 256, 0, stream>>>(
            out_w + l * HID * HID, wh, HID, HID);
        wmma_gemm_kernel<<<dim3(HID / 64, M / 64), 256, 0, stream>>>(
            ctxh, wh, out_b + l * HID, tmp, M, HID, HID, 0);
        add_ln_kernel<<<M, 256, 0, stream>>>(xbuf, tmp, ln1_g + l * HID, ln1_b + l * HID, xbuf, xh);
        // FFN + LN2
        f2h_transpose_kernel<<<(HID * 4 * HID + 255) / 256, 256, 0, stream>>>(
            ffn_w1 + l * HID * 4 * HID, wh, HID, 4 * HID);
        wmma_gemm_kernel<<<dim3(4 * HID / 64, M / 64), 256, 0, stream>>>(
            xh, wh, ffn_b1 + l * 4 * HID, h1, M, 4 * HID, HID, 1);
        f2h4_kernel<<<(M * 4 * HID / 4 + 255) / 256, 256, 0, stream>>>(h1, h1h, M * 4 * HID / 4);
        f2h_transpose_kernel<<<(4 * HID * HID + 255) / 256, 256, 0, stream>>>(
            ffn_w2 + l * 4 * HID * HID, wh, 4 * HID, HID);
        wmma_gemm_kernel<<<dim3(HID / 64, M / 64), 256, 0, stream>>>(
            h1h, wh, ffn_b2 + l * HID, tmp, M, HID, 4 * HID, 0);
        add_ln_kernel<<<M, 256, 0, stream>>>(xbuf, tmp, ln2_g + l * HID, ln2_b + l * HID, xbuf, xh);
    }

    candi_fused_kernel<<<M, 256, 0, stream>>>(xbuf, src_len, candi_ids, candi_feats,
                                              candi_masks, emb, fcw, fcb, fusion_g, fusion_b,
                                              pool, outF, outMM);
    pool_gate_kernel<<<BSZ, 256, 0, stream>>>(pool, src_len, pro, temporal_w,
                                              fc_hid_w, fc_hid_b, outHid);
}
